// HoloSpectralBlock_52561809768537
// MI455X (gfx1250) — compile-verified, matrix operand order checked by reading, not run on hardware
//
#include <hip/hip_runtime.h>
#include <hip/hip_bf16.h>
#include <stdint.h>

// ---------------------------------------------------------------------------
// HoloSpectralBlock for MI455X (gfx1250), wave32 + WMMA f16->f32.
//   B=4, T=2048, D=1024, HIDDEN=2048, M = B*T = 8192 tokens.
//
//   All GEMM B-operands are pre-laid-out COLUMN-MAJOR (N x K, f16) so both
//   A and B LDS staging are b128 loads + b128 stores (no transpose scatter).
//   GEMM: 128x128 block tile, BK=32, 8 waves x (32x64) = 8 WMMA/wave/K-step.
// ---------------------------------------------------------------------------

#define BB 4
#define TT 2048
#define DD 1024
#define HH 2048
#define MM (BB*TT)          // 8192
#define CHUNK 32

#define BM 128
#define BN 128
#define BK 32

typedef __attribute__((ext_vector_type(16))) _Float16 v16h;
typedef __attribute__((ext_vector_type(8)))  float    v8f;
typedef __attribute__((ext_vector_type(4)))  _Float16 h4;

__device__ __forceinline__ float sigmoidf_(float v) { return 1.0f / (1.0f + __expf(-v)); }

// ---------------- K0a: circulant matrices, column-major (n-major) ----------
__global__ __launch_bounds__(256)
void circulant_kernel(const float* __restrict__ bk, const float* __restrict__ uk,
                      _Float16* __restrict__ Cb, _Float16* __restrict__ Cc) {
  int idx = blockIdx.x * 256 + threadIdx.x;       // D*D = 1M, idx = n*D + k
  int n = idx >> 10, k = idx & 1023;
  Cb[idx] = (_Float16)bk[(n - k) & 1023];         // bound[.,n] += xn[.,k]*bk[(n-k)%D]
  Cc[idx] = (_Float16)uk[(k - n) & 1023];         // retr [.,n] += h [.,k]*uk[(k-n)%D]
}

// ---------------- K0b: plain f32 -> f16 convert (layout already N x K) -----
__global__ __launch_bounds__(256)
void convert_f16_kernel(const float* __restrict__ src, _Float16* __restrict__ dst,
                        long n) {
  long idx = (long)blockIdx.x * 256 + threadIdx.x;
  if (idx < n) dst[idx] = (_Float16)src[idx];
}

// ---------------- K0c: spectral 32x32 real operator ------------------------
__global__ __launch_bounds__(256)
void spectral_w_kernel(const float* __restrict__ fg, float* __restrict__ W) {
  int idx = blockIdx.x * 256 + threadIdx.x;       // 1024
  if (idx >= 1024) return;
  int n = idx >> 5, m = idx & 31;
  float g0  = sigmoidf_(fg[0]);
  float g16 = sigmoidf_(fg[16]);
  float acc = g0 + (((n - m) & 1) ? -g16 : g16);
  for (int f = 1; f < 16; ++f) {
    float gf = sigmoidf_(fg[f]);
    acc += 2.0f * gf * cosf(6.283185307179586f * (float)f * (float)(n - m) * (1.0f / 32.0f));
  }
  W[idx] = acc * (1.0f / 32.0f);
}

// ------------------------------- LN kernel ---------------------------------
__global__ __launch_bounds__(256)
void ln_kernel(const float* __restrict__ x, const float* __restrict__ w,
               const float* __restrict__ b, float* __restrict__ of32,
               _Float16* __restrict__ of16) {
  const int row = blockIdx.x;
  const int tid = threadIdx.x;
  const float4* xr4 = (const float4*)(x + (long)row * DD);
  float4 v = xr4[tid];                            // 16B/lane, fully coalesced
  float s  = v.x + v.y + v.z + v.w;
  float ss = v.x*v.x + v.y*v.y + v.z*v.z + v.w*v.w;
#pragma unroll
  for (int o = 16; o > 0; o >>= 1) { s += __shfl_xor(s, o, 32); ss += __shfl_xor(ss, o, 32); }
  __shared__ float rs8[8], rss8[8];
  if ((tid & 31) == 0) { rs8[tid >> 5] = s; rss8[tid >> 5] = ss; }
  __syncthreads();
  float ts = 0.f, tss = 0.f;
#pragma unroll
  for (int i = 0; i < 8; ++i) { ts += rs8[i]; tss += rss8[i]; }
  float mu = ts * (1.0f / DD);
  float rsig = rsqrtf(tss * (1.0f / DD) - mu * mu + 1e-5f);
  const float4* w4 = (const float4*)w;
  const float4* b4 = (const float4*)b;
  float4 wv = w4[tid], bv = b4[tid];
  float4 y;
  y.x = (v.x - mu) * rsig * wv.x + bv.x;
  y.y = (v.y - mu) * rsig * wv.y + bv.y;
  y.z = (v.z - mu) * rsig * wv.z + bv.z;
  y.w = (v.w - mu) * rsig * wv.w + bv.w;
  if (of32) ((float4*)(of32 + (long)row * DD))[tid] = y;
  if (of16) {
    h4 p; p.x = (_Float16)y.x; p.y = (_Float16)y.y; p.z = (_Float16)y.z; p.w = (_Float16)y.w;
    ((h4*)(of16 + (long)row * DD))[tid] = p;
  }
}

// -------------------------- EMA scan (decay mix) ---------------------------
__global__ __launch_bounds__(256)
void scan_kernel(const float* __restrict__ bound, _Float16* __restrict__ holo,
                 const float* __restrict__ dptr) {
  int t = blockIdx.x * 256 + threadIdx.x;         // B*D = 4096 channels
  int b = t >> 10, d = t & 1023;
  float dec = sigmoidf_(dptr[0]);
  const float* src = bound + ((size_t)b * TT) * DD + d;
  _Float16*    dst = holo  + ((size_t)b * TT) * DD + d;
  float h = 0.f;
  for (int s = 0; s < TT; s += 4) {               // batch loads ahead of serial FMAs
    float b0 = src[(size_t)(s + 0) * DD];
    float b1 = src[(size_t)(s + 1) * DD];
    float b2 = src[(size_t)(s + 2) * DD];
    float b3 = src[(size_t)(s + 3) * DD];
    h = b0 + dec * h; dst[(size_t)(s + 0) * DD] = (_Float16)h;
    h = b1 + dec * h; dst[(size_t)(s + 1) * DD] = (_Float16)h;
    h = b2 + dec * h; dst[(size_t)(s + 2) * DD] = (_Float16)h;
    h = b3 + dec * h; dst[(size_t)(s + 3) * DD] = (_Float16)h;
  }
}

// ------------------------- WMMA GEMM (f16 -> f32) --------------------------
// C[M,N] = A[M,K] @ B[K,N], with B given COLUMN-MAJOR as Bt[N,K].
// MODE 0: Cf32 = acc
// MODE 1: Cf32 = resid + extra + gptr[0]*acc
// MODE 2: Cf16 = gelu(acc)
// MODE 3: Cf32 = resid + svec[col]*acc
template <int MODE>
__global__ __launch_bounds__(256)
void wmma_gemm(const _Float16* __restrict__ A, const _Float16* __restrict__ Bt,
               float* __restrict__ Cf32, _Float16* __restrict__ Cf16,
               const float* __restrict__ resid, const float* __restrict__ extra,
               const float* __restrict__ svec, const float* __restrict__ gptr,
               int Ndim, int Kdim) {
  __shared__ __align__(16) _Float16 lA[BM * BK];  // [m][k]
  __shared__ __align__(16) _Float16 lB[BN * BK];  // [n][k]
  const int tid  = threadIdx.x;
  const int lane = tid & 31, wave = tid >> 5;
  const int lo = lane & 15, hi = lane >> 4;
  const int wm = wave & 3;                        // 4 slots of 32 rows
  const int wn = wave >> 2;                       // 2 slots of 64 cols
  const int m0 = blockIdx.x * BM, n0 = blockIdx.y * BN;

  const uint4* A4 = (const uint4*)A;
  const uint4* B4 = (const uint4*)Bt;
  uint4* lA4 = (uint4*)lA;
  uint4* lB4 = (uint4*)lB;
  const uint32_t* lAu = (const uint32_t*)lA;
  const uint32_t* lBu = (const uint32_t*)lB;
  const int K8 = Kdim >> 3;                       // uint4 per row

  v8f acc[2][4];
  {
    v8f z = {};
#pragma unroll
    for (int mt = 0; mt < 2; ++mt)
#pragma unroll
      for (int nt = 0; nt < 4; ++nt) acc[mt][nt] = z;
  }

  const int r  = tid >> 2;                        // 0..63 tile row
  const int c4 = tid & 3;                         // 4 uint4 per 32-half row

  for (int k0 = 0; k0 < Kdim; k0 += BK) {
    const int kc = (k0 >> 3) + c4;
    // A tile 128x32 halves = 512 uint4, B tile same: b128 load + b128 store
    lA4[(r     ) * 4 + c4] = A4[(size_t)(m0 + r     ) * K8 + kc];
    lA4[(r + 64) * 4 + c4] = A4[(size_t)(m0 + r + 64) * K8 + kc];
    lB4[(r     ) * 4 + c4] = B4[(size_t)(n0 + r     ) * K8 + kc];
    lB4[(r + 64) * 4 + c4] = B4[(size_t)(n0 + r + 64) * K8 + kc];
    if (k0 + BK < Kdim) {                         // global_prefetch_b8 next tiles
      __builtin_prefetch(&A4[(size_t)(m0 + r) * K8 + kc + (BK >> 3)], 0, 1);
      __builtin_prefetch(&B4[(size_t)(n0 + r) * K8 + kc + (BK >> 3)], 0, 1);
    }
    __syncthreads();

    // fragments per CDNA5 16-bit A/B VGPR layout (CDNA5 ISA §7.12.2):
    // lane pair runs are K-contiguous -> two ds_load_b128 per fragment
    union Frag { v16h v; uint32_t u[8]; };
    Frag af[2], bf[4];
#pragma unroll
    for (int mt = 0; mt < 2; ++mt)
#pragma unroll
      for (int p = 0; p < 8; ++p) {
        int ku = p + ((p >= 4) ? 4 : 0) + hi * 4;
        af[mt].u[p] = lAu[(wm * 32 + mt * 16 + lo) * 16 + ku];
      }
#pragma unroll
    for (int nt = 0; nt < 4; ++nt)
#pragma unroll
      for (int p = 0; p < 8; ++p) {
        int ku = p + ((p >= 4) ? 4 : 0) + hi * 4;
        bf[nt].u[p] = lBu[(wn * 64 + nt * 16 + lo) * 16 + ku];
      }
#pragma unroll
    for (int mt = 0; mt < 2; ++mt)
#pragma unroll
      for (int nt = 0; nt < 4; ++nt)
        acc[mt][nt] = __builtin_amdgcn_wmma_f32_16x16x32_f16(
            false, af[mt].v, false, bf[nt].v, (short)0, acc[mt][nt], false, false);
    __syncthreads();
  }

  // epilogue: C/D layout -> row = j + 8*hi, col = lo (per 16x16 tile)
  float gate = (MODE == 1) ? gptr[0] : 0.0f;
#pragma unroll
  for (int mt = 0; mt < 2; ++mt)
#pragma unroll
    for (int nt = 0; nt < 4; ++nt) {
      int col = n0 + wn * 64 + nt * 16 + lo;
#pragma unroll
      for (int j = 0; j < 8; ++j) {
        int row = m0 + wm * 32 + mt * 16 + j + hi * 8;
        size_t off = (size_t)row * Ndim + col;
        float v = acc[mt][nt][j];
        if (MODE == 0) {
          Cf32[off] = v;
        } else if (MODE == 1) {
          Cf32[off] = resid[off] + extra[off] + gate * v;
        } else if (MODE == 2) {
          Cf16[off] = (_Float16)(0.5f * v * (1.0f + erff(v * 0.70710678118654752f)));
        } else {
          Cf32[off] = resid[off] + svec[col] * v;
        }
      }
    }
}

// ------------------ fused LN2 + spectral gate (per 32-chunk) ---------------
__global__ __launch_bounds__(256)
void spectral_kernel(const float* __restrict__ x1, const float* __restrict__ w,
                     const float* __restrict__ bb, const float* __restrict__ W,
                     float* __restrict__ x2) {
  __shared__ float Ws[1024];
  __shared__ float mu[32], rs[32];
  __shared__ float red[256], red2[256];
  const int tid = threadIdx.x;
  const long base = (long)blockIdx.x * CHUNK * DD;   // (b,chunk) tile start

  for (int i = tid; i < 1024; i += 256) Ws[i] = W[i];

  // per-token stats: 8 threads per token row
  int rrow = tid >> 3, sub = tid & 7;
  {
    const float* xr = x1 + base + (long)rrow * DD;
    float s = 0.f, ss = 0.f;
    for (int i = 0; i < 128; ++i) { float v = xr[sub + i * 8]; s += v; ss += v * v; }
    red[tid] = s; red2[tid] = ss;
  }
  __syncthreads();
  if (sub == 0) {
    float a = 0.f, c2 = 0.f;
#pragma unroll
    for (int i = 0; i < 8; ++i) { a += red[rrow * 8 + i]; c2 += red2[rrow * 8 + i]; }
    float m_ = a * (1.0f / DD);
    mu[rrow] = m_;
    rs[rrow] = rsqrtf(c2 * (1.0f / DD) - m_ * m_ + 1e-5f);
  }
  __syncthreads();

  for (int ft = 0; ft < 4; ++ft) {
    int d = ft * 256 + tid;
    float wv = w[d], bv = bb[d];
    float vv[32], xn[32];
#pragma unroll
    for (int m = 0; m < 32; ++m) {
      float v = x1[base + (long)m * DD + d];
      vv[m] = v;
      xn[m] = (v - mu[m]) * rs[m] * wv + bv;
    }
#pragma unroll 4
    for (int t = 0; t < 32; ++t) {
      float acc = 0.f;
#pragma unroll
      for (int m = 0; m < 32; ++m) acc += Ws[t * 32 + m] * xn[m];
      x2[base + (long)t * DD + d] = vv[t] + acc;
    }
  }
}

// ---------------------------------------------------------------------------
extern "C" void kernel_launch(void* const* d_in, const int* in_sizes, int n_in,
                              void* d_out, int out_size, void* d_ws, size_t ws_size,
                              hipStream_t stream) {
  (void)in_sizes; (void)n_in; (void)out_size; (void)ws_size;
  const float* x    = (const float*)d_in[0];
  const float* bk   = (const float*)d_in[1];
  const float* uk   = (const float*)d_in[2];
  const float* gate = (const float*)d_in[3];
  const float* dec  = (const float*)d_in[4];
  const float* fg   = (const float*)d_in[5];
  const float* ln1w = (const float*)d_in[6];
  const float* ln1b = (const float*)d_in[7];
  const float* ln2w = (const float*)d_in[8];
  const float* ln2b = (const float*)d_in[9];
  const float* ln3w = (const float*)d_in[10];
  const float* ln3b = (const float*)d_in[11];
  const float* dw   = (const float*)d_in[12];   // (HIDDEN, D)  == Bt for GEMM1
  const float* uw   = (const float*)d_in[13];   // (D, HIDDEN)  == Bt for GEMM2
  const float* sc   = (const float*)d_in[14];
  float* out = (float*)d_out;

  // workspace carve-up (256B aligned)
  char* w = (char*)d_ws;
  size_t o = 0;
  auto take = [&](size_t bytes) { size_t r = o; o += (bytes + 255) & ~(size_t)255; return r; };
  float*    xn1f  = (float*)   (w + take((size_t)MM * DD * 4));
  _Float16* xn1h  = (_Float16*)(w + take((size_t)MM * DD * 2));
  _Float16* Cbind = (_Float16*)(w + take((size_t)DD * DD * 2));   // col-major
  _Float16* Ccorr = (_Float16*)(w + take((size_t)DD * DD * 2));   // col-major
  float*    bound = (float*)   (w + take((size_t)MM * DD * 4));
  _Float16* holoh = (_Float16*)(w + take((size_t)MM * DD * 2));
  float*    x1    = (float*)   (w + take((size_t)MM * DD * 4));
  float*    x2    = (float*)   (w + take((size_t)MM * DD * 4));
  _Float16* xn3h  = (_Float16*)(w + take((size_t)MM * DD * 2));
  _Float16* Bdown = (_Float16*)(w + take((size_t)DD * HH * 2));   // col-major
  _Float16* Bup   = (_Float16*)(w + take((size_t)HH * DD * 2));   // col-major
  _Float16* hmid  = (_Float16*)(w + take((size_t)MM * HH * 2));
  float*    Wspec = (float*)   (w + take(1024 * 4));

  // K0: operand builders (one-time; (out,in) weight layout is already N x K)
  circulant_kernel<<<(DD * DD) / 256, 256, 0, stream>>>(bk, uk, Cbind, Ccorr);
  convert_f16_kernel<<<(HH * DD + 255) / 256, 256, 0, stream>>>(dw, Bdown, (long)HH * DD);
  convert_f16_kernel<<<(DD * HH + 255) / 256, 256, 0, stream>>>(uw, Bup, (long)DD * HH);
  spectral_w_kernel<<<4, 256, 0, stream>>>(fg, Wspec);

  // K1: LN1
  ln_kernel<<<MM, 256, 0, stream>>>(x, ln1w, ln1b, xn1f, xn1h);

  // K2: bound = xn1 @ Cbind
  wmma_gemm<0><<<dim3(MM / BM, DD / BN), 256, 0, stream>>>(
      xn1h, Cbind, bound, nullptr, nullptr, nullptr, nullptr, nullptr, DD, DD);

  // K3: causal EMA scan
  scan_kernel<<<(BB * DD) / 256, 256, 0, stream>>>(bound, holoh, dec);

  // K4: x1 = x + xn1 + gate * (holo @ Ccorr)
  wmma_gemm<1><<<dim3(MM / BM, DD / BN), 256, 0, stream>>>(
      holoh, Ccorr, x1, nullptr, x, xn1f, nullptr, gate, DD, DD);

  // K5: fused LN2 + spectral gate
  spectral_kernel<<<BB * (TT / CHUNK), 256, 0, stream>>>(x1, ln2w, ln2b, Wspec, x2);

  // K6: LN3 (f16 only)
  ln_kernel<<<MM, 256, 0, stream>>>(x2, ln3w, ln3b, nullptr, xn3h);

  // K7: hmid = GELU(xn3 @ down_w^T)  -> f16
  wmma_gemm<2><<<dim3(MM / BM, HH / BN), 256, 0, stream>>>(
      xn3h, Bdown, nullptr, hmid, nullptr, nullptr, nullptr, nullptr, HH, DD);

  // K8: out = x2 + scale * (hmid @ up_w^T)
  wmma_gemm<3><<<dim3(MM / BM, DD / BN), 256, 0, stream>>>(
      hmid, Bup, out, nullptr, x2, nullptr, sc, nullptr, DD, HH);
}
